// MultiHeadAttention_64802466562731
// MI455X (gfx1250) — compile-verified, matrix operand order checked by reading, not run on hardware
//
#include <hip/hip_runtime.h>
#include <hip/hip_bf16.h>
#include <math.h>
#include <stdint.h>

#define Bv 4
#define Sv 2048
#define Dv 768
#define Hv 12
#define HDv 64
#define BSv (Bv * Sv)   // 8192
#define KSTEPS (Dv / 32)  // 24

typedef __attribute__((ext_vector_type(16))) _Float16 v16h;
typedef __attribute__((ext_vector_type(8)))  _Float16 v8h;
typedef __attribute__((ext_vector_type(8)))  float    v8f;

// ---------------------------------------------------------------------------
// WMMA wrapper: D = A(16x32 f16) * B(32x16 f16) + C(16x16 f32)
// ---------------------------------------------------------------------------
__device__ __forceinline__ v8f wmma_f16(v16h a, v16h b, v8f c) {
  return __builtin_amdgcn_wmma_f32_16x16x32_f16(
      /*neg_a=*/false, a, /*neg_b=*/false, b,
      /*c_mod=*/(short)0, c, /*reuse_a=*/false, /*reuse_b=*/false);
}

// ---------------------------------------------------------------------------
// Fragment loader for the 16-bit A/B WMMA layout.
// Tile is 16 (row/col) x 32 (K); element [r][k] lives at base[r*stride + k].
// Lane L holds r = L&15, half h = L>>4:
//   v16h[0..7]  = K = h*8 + 0..7 ; v16h[8..15] = K = 16 + h*8 + 0..7
// ---------------------------------------------------------------------------
__device__ __forceinline__ v16h load_frag(const _Float16* __restrict__ base,
                                          int stride, int lane) {
  int r = lane & 15;
  int h = (lane >> 4) & 1;
  const _Float16* p = base + (size_t)r * stride + h * 8;
  v8h lo = *(const v8h*)(p);
  v8h hi = *(const v8h*)(p + 16);
  return __builtin_shufflevector(lo, hi, 0, 1, 2, 3, 4, 5, 6, 7,
                                         8, 9, 10, 11, 12, 13, 14, 15);
}

// Async global->LDS copy of 16 bytes per lane (gfx1250 ASYNCcnt path)
__device__ __forceinline__ void async_copy_b128(const _Float16* gsrc,
                                                _Float16* ldst) {
  unsigned loff = (unsigned)(uintptr_t)ldst;  // LDS byte offset = addr[31:0]
  asm volatile("global_load_async_to_lds_b128 %0, %1, off"
               :: "v"(loff), "v"(gsrc) : "memory");
}
__device__ __forceinline__ void wait_async0() {
  asm volatile("s_wait_asynccnt 0x0" ::: "memory");
}

// half-wave (16 lane) reductions: a C-tile row spans 16 lanes of one VGPR
__device__ __forceinline__ float hmax16(float v) {
  #pragma unroll
  for (int m = 1; m < 16; m <<= 1) v = fmaxf(v, __shfl_xor(v, m, 16));
  return v;
}
__device__ __forceinline__ float hsum16(float v) {
  #pragma unroll
  for (int m = 1; m < 16; m <<= 1) v += __shfl_xor(v, m, 16);
  return v;
}

// ---------------------------------------------------------------------------
// 1) fp32 -> f16 convert
// ---------------------------------------------------------------------------
__global__ void cvt_f32_f16(const float* __restrict__ src,
                            _Float16* __restrict__ dst, int n) {
  int i = blockIdx.x * blockDim.x + threadIdx.x;
  if (i < n) dst[i] = (_Float16)src[i];
}

// ---------------------------------------------------------------------------
// Shared GEMM mainloop: acc[8] (2 m-subtiles x 4 n-subtiles), K = 768.
// W tile (64 cols x 32 k) double-buffered in LDS via async-to-LDS copies
// (shared by all 8 waves of the block); A fragments software-pipelined
// one k-step ahead in registers.
// ---------------------------------------------------------------------------
__device__ __forceinline__ void gemm_mainloop(
    const _Float16* __restrict__ Xh, const _Float16* __restrict__ Wh,
    int m0, int n0, int lane, int tid, _Float16* btile /* [2][64*32] */,
    v8f acc[8]) {
  int r = tid >> 2;   // 0..63 : weight row (output col n0+r)
  int c = tid & 3;    // 0..3  : 8-half chunk within the 32-wide k slice
  const _Float16* wsrc = Wh + (size_t)(n0 + r) * Dv + c * 8;
  _Float16* ldst0 = btile + r * 32 + c * 8;   // buffer 0 slot
  _Float16* ldst1 = ldst0 + 64 * 32;          // buffer 1 slot

  const _Float16* arow0 = Xh + (size_t)m0 * Dv;
  const _Float16* arow1 = Xh + (size_t)(m0 + 16) * Dv;

  async_copy_b128(wsrc, ldst0);               // stage k-step 0 into buffer 0
  v16h a0 = load_frag(arow0, Dv, lane);       // preload A for k-step 0
  v16h a1 = load_frag(arow1, Dv, lane);

  for (int kk = 0; kk < KSTEPS; ++kk) {
    int cur = kk & 1;
    wait_async0();
    __syncthreads();
    if (kk + 1 < KSTEPS)  // prefetch next W tile into the other buffer
      async_copy_b128(wsrc + (size_t)(kk + 1) * 32, (cur == 0) ? ldst1 : ldst0);

    const _Float16* bt = btile + cur * (64 * 32);
    #pragma unroll
    for (int t = 0; t < 4; ++t) {
      v16h b = load_frag(bt + t * 16 * 32, 32, lane);
      acc[t]     = wmma_f16(a0, b, acc[t]);
      acc[4 + t] = wmma_f16(a1, b, acc[4 + t]);
    }
    if (kk + 1 < KSTEPS) {  // pipeline A fragments one step ahead
      a0 = load_frag(arow0 + (kk + 1) * 32, Dv, lane);
      a1 = load_frag(arow1 + (kk + 1) * 32, Dv, lane);
    }
  }
}

// ---------------------------------------------------------------------------
// 2) QKV projection: out = X @ W^T   (M=8192, N=768, K=768)
// mode 0: store [b][h][s][hd]   (Q, K)
// mode 1: store [b][h][hd][s]   (V transposed, keys contiguous)
// grid: (32, 12), block 256 (8 waves); wave -> 32x64 output tile
// ---------------------------------------------------------------------------
__global__ __launch_bounds__(256) void proj_kernel(
    const _Float16* __restrict__ Xh, const _Float16* __restrict__ Wh,
    _Float16* __restrict__ out, int mode) {
  __shared__ __align__(16) _Float16 btile[2][64 * 32];
  int tid = threadIdx.x;
  int wave = tid >> 5, lane = tid & 31;
  int m0 = blockIdx.x * 256 + wave * 32;
  int n0 = blockIdx.y * 64;

  v8f acc[8];
  #pragma unroll
  for (int t = 0; t < 8; ++t) acc[t] = (v8f){0, 0, 0, 0, 0, 0, 0, 0};

  gemm_mainloop(Xh, Wh, m0, n0, lane, tid, &btile[0][0], acc);

  int ncol = lane & 15;
  int rowbase = (lane >> 4) * 8;
  #pragma unroll
  for (int mi = 0; mi < 2; ++mi) {
    #pragma unroll
    for (int t = 0; t < 4; ++t) {
      #pragma unroll
      for (int j = 0; j < 8; ++j) {
        int m = m0 + mi * 16 + rowbase + j;
        int n = n0 + t * 16 + ncol;
        int bb = m >> 11;
        int s = m & (Sv - 1);
        int head = n >> 6;
        int hd = n & (HDv - 1);
        _Float16 val = (_Float16)acc[mi * 4 + t][j];
        if (mode == 0)
          out[(((size_t)bb * Hv + head) * Sv + s) * HDv + hd] = val;
        else
          out[(((size_t)bb * Hv + head) * HDv + hd) * Sv + s] = val;
      }
    }
  }
}

// ---------------------------------------------------------------------------
// 3) Flash attention (causal). grid: (S/128, B*H), block 256 (8 waves).
// Each wave owns 16 query rows (Q frags resident in registers); key blocks
// of 32; online softmax with half-wave shuffles; P relayout via LDS.
// ---------------------------------------------------------------------------
__global__ __launch_bounds__(256) void attn_kernel(
    const _Float16* __restrict__ Qh, const _Float16* __restrict__ Kh,
    const _Float16* __restrict__ Vth, _Float16* __restrict__ ctxh) {
  __shared__ __align__(16) _Float16 Plds[8][16 * 32];

  int wave = threadIdx.x >> 5;
  int lane = threadIdx.x & 31;
  int bh = blockIdx.y;
  int b = bh / Hv, h = bh % Hv;
  int q0 = blockIdx.x * 128 + wave * 16;

  const _Float16* Qbase  = Qh  + ((size_t)(b * Hv + h) * Sv + q0) * HDv;
  const _Float16* Kbase  = Kh  + ((size_t)(b * Hv + h) * Sv) * HDv;
  const _Float16* Vtbase = Vth + ((size_t)(b * Hv + h) * HDv) * Sv;

  v16h qa0 = load_frag(Qbase + 0,  HDv, lane);
  v16h qa1 = load_frag(Qbase + 32, HDv, lane);

  v8f o0 = {0,0,0,0,0,0,0,0}, o1 = {0,0,0,0,0,0,0,0};
  v8f o2 = {0,0,0,0,0,0,0,0}, o3 = {0,0,0,0,0,0,0,0};
  float mrow[8], lrow[8];
  #pragma unroll
  for (int j = 0; j < 8; ++j) { mrow[j] = -INFINITY; lrow[j] = 0.0f; }

  int ncol = lane & 15;
  int rowbase = (lane >> 4) * 8;
  const float scale = 0.125f;  // 1/sqrt(64)

  for (int kb = 0; kb <= q0 + 15; kb += 32) {
    // prefetch next K block (global_prefetch_b8); 32 lanes x 128B = 4KB tile
    if (kb + 32 <= q0 + 15)
      __builtin_prefetch(Kbase + (size_t)(kb + 32) * HDv + lane * 64, 0, 3);

    // ---- scores: two 16-key tiles, HD split into 2 K-chunks of 32 ----
    v8f c0 = {0,0,0,0,0,0,0,0}, c1 = {0,0,0,0,0,0,0,0};
    {
      v16h bk;
      bk = load_frag(Kbase + (size_t)kb * HDv + 0,  HDv, lane);
      c0 = wmma_f16(qa0, bk, c0);
      bk = load_frag(Kbase + (size_t)kb * HDv + 32, HDv, lane);
      c0 = wmma_f16(qa1, bk, c0);
      bk = load_frag(Kbase + (size_t)(kb + 16) * HDv + 0,  HDv, lane);
      c1 = wmma_f16(qa0, bk, c1);
      bk = load_frag(Kbase + (size_t)(kb + 16) * HDv + 32, HDv, lane);
      c1 = wmma_f16(qa1, bk, c1);
    }

    // ---- causal mask + scale + online softmax ----
    #pragma unroll
    for (int j = 0; j < 8; ++j) {
      int q = q0 + rowbase + j;
      float s0 = (kb + ncol      > q) ? -INFINITY : c0[j] * scale;
      float s1 = (kb + 16 + ncol > q) ? -INFINITY : c1[j] * scale;
      float mt = hmax16(fmaxf(s0, s1));
      float mn = fmaxf(mrow[j], mt);
      float alpha = __expf(mrow[j] - mn);
      float p0 = __expf(s0 - mn);
      float p1 = __expf(s1 - mn);
      lrow[j] = lrow[j] * alpha + hsum16(p0 + p1);
      mrow[j] = mn;
      o0[j] *= alpha; o1[j] *= alpha; o2[j] *= alpha; o3[j] *= alpha;
      Plds[wave][(rowbase + j) * 32 + ncol]      = (_Float16)p0;
      Plds[wave][(rowbase + j) * 32 + 16 + ncol] = (_Float16)p1;
    }

    // ---- P (16x32) * V (32x64): reload P in A-fragment layout ----
    v16h pa = load_frag(&Plds[wave][0], 32, lane);
    o0 = wmma_f16(pa, load_frag(Vtbase + (size_t)(0 * 16) * Sv + kb, Sv, lane), o0);
    o1 = wmma_f16(pa, load_frag(Vtbase + (size_t)(1 * 16) * Sv + kb, Sv, lane), o1);
    o2 = wmma_f16(pa, load_frag(Vtbase + (size_t)(2 * 16) * Sv + kb, Sv, lane), o2);
    o3 = wmma_f16(pa, load_frag(Vtbase + (size_t)(3 * 16) * Sv + kb, Sv, lane), o3);
  }

  #pragma unroll
  for (int j = 0; j < 8; ++j) {
    float inv = 1.0f / lrow[j];
    int s = q0 + rowbase + j;
    size_t base = ((size_t)b * Sv + s) * Dv + h * HDv;
    ctxh[base + 0 * 16 + ncol] = (_Float16)(o0[j] * inv);
    ctxh[base + 1 * 16 + ncol] = (_Float16)(o1[j] * inv);
    ctxh[base + 2 * 16 + ncol] = (_Float16)(o2[j] * inv);
    ctxh[base + 3 * 16 + ncol] = (_Float16)(o3[j] * inv);
  }
}

// ---------------------------------------------------------------------------
// 4) Output projection: out = ctx @ Wo^T + bo (f32 out)
// ---------------------------------------------------------------------------
__global__ __launch_bounds__(256) void oproj_kernel(
    const _Float16* __restrict__ Ch, const _Float16* __restrict__ Woh,
    const float* __restrict__ bo, float* __restrict__ out) {
  __shared__ __align__(16) _Float16 btile[2][64 * 32];
  int tid = threadIdx.x;
  int wave = tid >> 5, lane = tid & 31;
  int m0 = blockIdx.x * 256 + wave * 32;
  int n0 = blockIdx.y * 64;

  v8f acc[8];
  #pragma unroll
  for (int t = 0; t < 8; ++t) acc[t] = (v8f){0, 0, 0, 0, 0, 0, 0, 0};

  gemm_mainloop(Ch, Woh, m0, n0, lane, tid, &btile[0][0], acc);

  int ncol = lane & 15;
  int rowbase = (lane >> 4) * 8;
  #pragma unroll
  for (int mi = 0; mi < 2; ++mi) {
    #pragma unroll
    for (int t = 0; t < 4; ++t) {
      float bias = bo[n0 + t * 16 + ncol];
      #pragma unroll
      for (int j = 0; j < 8; ++j) {
        int m = m0 + mi * 16 + rowbase + j;
        int n = n0 + t * 16 + ncol;
        out[(size_t)m * Dv + n] = acc[mi * 4 + t][j] + bias;
      }
    }
  }
}

// ---------------------------------------------------------------------------
// Host launcher
// ---------------------------------------------------------------------------
extern "C" void kernel_launch(void* const* d_in, const int* in_sizes, int n_in,
                              void* d_out, int out_size, void* d_ws, size_t ws_size,
                              hipStream_t stream) {
  const float* x  = (const float*)d_in[0];
  const float* wq = (const float*)d_in[1];
  const float* wk = (const float*)d_in[2];
  const float* wv = (const float*)d_in[3];
  const float* wo = (const float*)d_in[4];
  const float* bo = (const float*)d_in[5];
  float* out = (float*)d_out;

  const size_t NX = (size_t)BSv * Dv;   // 6291456
  const size_t NW = (size_t)Dv * Dv;    // 589824

  _Float16* ws = (_Float16*)d_ws;
  _Float16* xh   = ws;              ws += NX;
  _Float16* wqh  = ws;              ws += NW;
  _Float16* wkh  = ws;              ws += NW;
  _Float16* wvh  = ws;              ws += NW;
  _Float16* woh  = ws;              ws += NW;
  _Float16* Qh   = ws;              ws += NX;
  _Float16* Kh   = ws;              ws += NX;
  _Float16* Vth  = ws;              ws += NX;
  _Float16* ctxh = ws;              ws += NX;

  cvt_f32_f16<<<(int)((NX + 255) / 256), 256, 0, stream>>>(x, xh, (int)NX);
  cvt_f32_f16<<<(int)((NW + 255) / 256), 256, 0, stream>>>(wq, wqh, (int)NW);
  cvt_f32_f16<<<(int)((NW + 255) / 256), 256, 0, stream>>>(wk, wkh, (int)NW);
  cvt_f32_f16<<<(int)((NW + 255) / 256), 256, 0, stream>>>(wv, wvh, (int)NW);
  cvt_f32_f16<<<(int)((NW + 255) / 256), 256, 0, stream>>>(wo, woh, (int)NW);

  dim3 pgrid(BSv / 256, Dv / 64);   // (32, 12)
  proj_kernel<<<pgrid, 256, 0, stream>>>(xh, wqh, Qh, 0);
  proj_kernel<<<pgrid, 256, 0, stream>>>(xh, wkh, Kh, 0);
  proj_kernel<<<pgrid, 256, 0, stream>>>(xh, wvh, Vth, 1);

  dim3 agrid(Sv / 128, Bv * Hv);    // (16, 48)
  attn_kernel<<<agrid, 256, 0, stream>>>(Qh, Kh, Vth, ctxh);

  oproj_kernel<<<pgrid, 256, 0, stream>>>(ctxh, woh, bo, out);
}
